// Attention_70909910057652
// MI455X (gfx1250) — matrix operand order repaired and checked by reading, not compiled, not run on hardware
//
#include <hip/hip_runtime.h>
#include <hip/hip_bf16.h>

// ---------------------------------------------------------------------------
// XCA (channel cross-covariance attention) for MI455X / gfx1250.
// All GEMMs use v_wmma_f32_16x16x32_bf16 (bf16 inputs, f32 accumulate).
// attn@v + proj are algebraically fused into a single GEMM to save ~200MB of
// HBM traffic on a bandwidth-bound pipeline.
// ---------------------------------------------------------------------------

typedef __attribute__((ext_vector_type(16))) __bf16 v16bf;
typedef __attribute__((ext_vector_type(8)))  __bf16 v8bf;
typedef __attribute__((ext_vector_type(8)))  float  v8f;

static __device__ __forceinline__ v16bf join16(v8bf lo, v8bf hi) {
  return __builtin_shufflevector(lo, hi, 0,1,2,3,4,5,6,7,8,9,10,11,12,13,14,15);
}

#define WMMA_BF16(A, B, C) \
  __builtin_amdgcn_wmma_f32_16x16x32_bf16(false, (A), false, (B), (short)0, (C), false, false)

// ============================= GEMM ========================================
// C[M][NT] (f32) = A[M][KT] (f32, row-major) x B[KT][NT] (f32, row-major)
// NT, KT compile-time so address math folds to immediates.
// Block tile 128x128, BK=32, double-buffered LDS (one barrier per K-step).
// 256 threads = 8 wave32: wave grid 2(M) x 4(N), each wave 64x32 = 4x2 frags.
#define BM 128
#define BN 128
#define BK 32
#define KPAD 8
#define KS (BK + KPAD)   // 40 bf16 = 80 bytes (16B-aligned row stride)

template <int NT, int KT>
__global__ __launch_bounds__(256)
void gemm_bf16_wmma(const float* __restrict__ A, long long a_bstride,
                    const float* __restrict__ B, long long b_bstride,
                    float* __restrict__ C, long long c_bstride)
{
  __shared__ __align__(16) __bf16 As[2][BM][KS];
  __shared__ __align__(16) __bf16 Bs[2][BN][KS];   // transposed: Bs[buf][n][k]

  const int bz = blockIdx.z;
  A += (long long)bz * a_bstride;
  B += (long long)bz * b_bstride;
  C += (long long)bz * c_bstride;

  const int row0 = blockIdx.y * BM;
  const int col0 = blockIdx.x * BN;

  const int tid  = threadIdx.x;
  const int lane = tid & 31;
  const int wave = tid >> 5;
  const int wm   = (wave >> 2) * 64;   // 0 / 64
  const int wn   = (wave & 3) * 32;    // 0,32,64,96

  // staging maps
  const int ar = tid >> 1;              // A: row 0..127
  const int ak = (tid & 1) * 16;        // A: k sub-block 0/16
  const int bn = tid & 127;             // B: column 0..127 (lane-consecutive)
  const int bq = (tid >> 7) * 16;       // B: k sub-block 0/16

  const float* agp = A + (long long)(row0 + ar) * KT + ak;
  const float* bgp = B + (long long)bq * NT + (col0 + bn);

  auto stage = [&](int buf, int k0) {
    // ---- A tile: 16 contiguous f32 -> two packed b128 LDS stores ----------
    {
      const float* gp = agp + k0;
#pragma unroll
      for (int u = 0; u < 16; u += 8) {
        float4 f0 = *(const float4*)(gp + u);
        float4 f1 = *(const float4*)(gp + u + 4);
        v8bf pk;
        pk[0] = (__bf16)f0.x; pk[1] = (__bf16)f0.y; pk[2] = (__bf16)f0.z; pk[3] = (__bf16)f0.w;
        pk[4] = (__bf16)f1.x; pk[5] = (__bf16)f1.y; pk[6] = (__bf16)f1.z; pk[7] = (__bf16)f1.w;
        *(v8bf*)&As[buf][ar][ak + u] = pk;
      }
    }
    // ---- B tile transposed: each thread reads 16 rows of its column -------
    {
      const float* gp = bgp + (long long)k0 * NT;
      float v[16];
#pragma unroll
      for (int j = 0; j < 16; ++j) v[j] = gp[j * NT];   // coalesced across lanes
#pragma unroll
      for (int u = 0; u < 16; u += 8) {
        v8bf pk;
#pragma unroll
        for (int e = 0; e < 8; ++e) pk[e] = (__bf16)v[u + e];
        *(v8bf*)&Bs[buf][bn][bq + u] = pk;
      }
    }
  };

  const v8f vzero = {0.f,0.f,0.f,0.f,0.f,0.f,0.f,0.f};
  v8f acc[4][2];
#pragma unroll
  for (int mi = 0; mi < 4; ++mi)
#pragma unroll
    for (int ni = 0; ni < 2; ++ni) acc[mi][ni] = vzero;

  const int half = lane >> 4;     // 0/1 lane half
  const int r16  = lane & 15;

  stage(0, 0);
  int p = 0;
  for (int k0 = 0; k0 < KT; k0 += BK) {
    __syncthreads();
    if (k0 + BK < KT) stage(p ^ 1, k0 + BK);   // overlap next-tile loads w/ WMMA

    v16bf af[4], bf[2];
#pragma unroll
    for (int mi = 0; mi < 4; ++mi) {
      // A 16x32 layout: lane half h holds K = h*8+{0..7} and 16+h*8+{0..7}
      const __bf16* q = &As[p][wm + mi*16 + r16][half * 8];
      af[mi] = join16(*(const v8bf*)q, *(const v8bf*)(q + 16));
    }
#pragma unroll
    for (int ni = 0; ni < 2; ++ni) {
      // B 32x16 layout: lane half h holds K = h*8+{0..7} and 16+h*8+{0..7}
      const __bf16* q = &Bs[p][wn + ni*16 + r16][half * 8];
      bf[ni] = join16(*(const v8bf*)q, *(const v8bf*)(q + 16));
    }
#pragma unroll
    for (int mi = 0; mi < 4; ++mi)
#pragma unroll
      for (int ni = 0; ni < 2; ++ni)
        acc[mi][ni] = WMMA_BF16(af[mi], bf[ni], acc[mi][ni]);
    p ^= 1;
  }

  // ---- epilogue: D layout — v[i]: lanes<16 -> M=i, lanes>=16 -> M=8+i ------
  const int mo = half * 8;
#pragma unroll
  for (int mi = 0; mi < 4; ++mi)
#pragma unroll
    for (int ni = 0; ni < 2; ++ni) {
      float* cp = C + (long long)(row0 + wm + mi*16 + mo) * NT + (col0 + wn + ni*16 + r16);
#pragma unroll
      for (int i = 0; i < 8; ++i) cp[i * NT] = acc[mi][ni][i];   // imm offsets
    }
}

// ===================== depthwise 3x3 + sum-of-squares ======================
__global__ __launch_bounds__(256)
void dwconv3x3_ssq(const float* __restrict__ in,   // [B*3C][H][W]
                   const float* __restrict__ w,    // [3C][9]
                   float* __restrict__ out,
                   float* __restrict__ ssq,        // [B][2C] (pre-zeroed)
                   int C3, int C, int H, int W)
{
  __shared__ float red[256];
  const int bc = blockIdx.z;            // b*3C + ch
  const int ch = bc % C3;
  const int b  = bc / C3;
  const int x  = blockIdx.x * 16 + (threadIdx.x & 15);
  const int y  = blockIdx.y * 16 + (threadIdx.x >> 4);

  const float* ip = in + (long long)bc * H * W;
  const float* wp = w + (long long)ch * 9;
  float wr[9];
#pragma unroll
  for (int i = 0; i < 9; ++i) wr[i] = wp[i];

  float acc = 0.f;
#pragma unroll
  for (int dy = 0; dy < 3; ++dy)
#pragma unroll
    for (int dx = 0; dx < 3; ++dx) {
      int yy = y + dy - 1, xx = x + dx - 1;
      if (yy >= 0 && yy < H && xx >= 0 && xx < W)
        acc = fmaf(wr[dy*3+dx], ip[(long long)yy * W + xx], acc);
    }
  out[(long long)bc * H * W + (long long)y * W + x] = acc;

  // accumulate sum(v^2) over spatial for q and k channels (L2 norm over N)
  if (ch < 2 * C) {
    red[threadIdx.x] = acc * acc;
    __syncthreads();
#pragma unroll
    for (int s = 128; s > 0; s >>= 1) {
      if (threadIdx.x < s) red[threadIdx.x] += red[threadIdx.x + s];
      __syncthreads();
    }
    if (threadIdx.x == 0)
      atomicAdd(&ssq[(long long)b * 2 * C + ch], red[0]);
  }
}

// ========================= attention: softmax(qn kn^T * T) =================
// One block per (b, head). q,k rows pre-scaled by 1/max(||row||,eps) while
// converting to bf16. M padded 48->64 (zero rows, filled once). K=16384 split
// across 8 waves per 256-column chunk; cross-wave reduce via ds_add_f32.
#define ACH 48
#define AM  64
#define AKC 256
#define AKS (AKC + 8)   // 264 bf16 = 528 B row stride (16B-aligned)

__global__ __launch_bounds__(256)
void attn_softmax_wmma(const float* __restrict__ qkv_dw,     // [B][3C][NN]
                       const float* __restrict__ ssq,        // [B][2C]
                       const float* __restrict__ temperature,// [8]
                       float* __restrict__ attn_out,         // [B][H][48][48]
                       int C, int NN)
{
  __shared__ __align__(16) __bf16 Qs[AM][AKS];
  __shared__ __align__(16) __bf16 Ks[AM][AKS];
  __shared__ float Ssum[AM][AM];

  const int bh = blockIdx.x;
  const int b = bh >> 3, h = bh & 7;
  const int ch0 = h * ACH;
  const float* qbase = qkv_dw + (long long)b * 3 * C * NN;
  const float* kbase = qbase + (long long)C * NN;
  const float* ssqb  = ssq + (long long)b * 2 * C;

  const int tid  = threadIdx.x;
  const int lane = tid & 31;
  const int wave = tid >> 5;
  const int half = lane >> 4;
  const int r16  = lane & 15;

  for (int i = tid; i < AM * AM; i += 256) (&Ssum[0][0])[i] = 0.f;

  const v8f vzero = {0.f,0.f,0.f,0.f,0.f,0.f,0.f,0.f};
  v8f acc[4][4];
#pragma unroll
  for (int mi = 0; mi < 4; ++mi)
#pragma unroll
    for (int ni = 0; ni < 4; ++ni) acc[mi][ni] = vzero;

  // staging map: thread -> (row 0..63, 64-column segment)
  const int sr   = tid >> 2;
  const int cseg = (tid & 3) * 64;
  float rq = 0.f, rk = 0.f;
  const float *gq = qbase, *gk = kbase;
  if (sr < ACH) {
    rq = 1.f / fmaxf(sqrtf(ssqb[ch0 + sr]), 1e-12f);
    rk = 1.f / fmaxf(sqrtf(ssqb[C + ch0 + sr]), 1e-12f);
    gq = qbase + (long long)(ch0 + sr) * NN + cseg;
    gk = kbase + (long long)(ch0 + sr) * NN + cseg;
  } else {
    // zero-pad rows 48..63 once; never rewritten inside the K loop
    v8bf zv;
#pragma unroll
    for (int e = 0; e < 8; ++e) zv[e] = (__bf16)0.f;
#pragma unroll
    for (int u = 0; u < 64; u += 8) {
      *(v8bf*)&Qs[sr][cseg + u] = zv;
      *(v8bf*)&Ks[sr][cseg + u] = zv;
    }
  }

  for (int k0 = 0; k0 < NN; k0 += AKC) {
    // ---- stage q,k chunk (scaled, bf16) as packed b128 stores -------------
    if (sr < ACH) {
#pragma unroll 2
      for (int u = 0; u < 64; u += 8) {
        float4 a0 = *(const float4*)(gq + k0 + u);
        float4 a1 = *(const float4*)(gq + k0 + u + 4);
        float4 b0 = *(const float4*)(gk + k0 + u);
        float4 b1 = *(const float4*)(gk + k0 + u + 4);
        v8bf pq, pk;
        pq[0]=(__bf16)(a0.x*rq); pq[1]=(__bf16)(a0.y*rq); pq[2]=(__bf16)(a0.z*rq); pq[3]=(__bf16)(a0.w*rq);
        pq[4]=(__bf16)(a1.x*rq); pq[5]=(__bf16)(a1.y*rq); pq[6]=(__bf16)(a1.z*rq); pq[7]=(__bf16)(a1.w*rq);
        pk[0]=(__bf16)(b0.x*rk); pk[1]=(__bf16)(b0.y*rk); pk[2]=(__bf16)(b0.z*rk); pk[3]=(__bf16)(b0.w*rk);
        pk[4]=(__bf16)(b1.x*rk); pk[5]=(__bf16)(b1.y*rk); pk[6]=(__bf16)(b1.z*rk); pk[7]=(__bf16)(b1.w*rk);
        *(v8bf*)&Qs[sr][cseg + u] = pq;
        *(v8bf*)&Ks[sr][cseg + u] = pk;
      }
    }
    __syncthreads();

    // ---- each wave: its 32-wide K slice of the chunk ---------------------
    const int ks = wave * 32;
    v16bf qf[4], kf[4];
#pragma unroll
    for (int mi = 0; mi < 4; ++mi) {
      const __bf16* q = &Qs[mi*16 + r16][ks + half*8];
      qf[mi] = join16(*(const v8bf*)q, *(const v8bf*)(q + 16));
    }
#pragma unroll
    for (int ni = 0; ni < 4; ++ni) {
      const __bf16* q = &Ks[ni*16 + r16][ks + half*8];
      kf[ni] = join16(*(const v8bf*)q, *(const v8bf*)(q + 16));
    }
#pragma unroll
    for (int mi = 0; mi < 4; ++mi)
#pragma unroll
      for (int ni = 0; ni < 4; ++ni)
        acc[mi][ni] = WMMA_BF16(qf[mi], kf[ni], acc[mi][ni]);
    __syncthreads();
  }

  // ---- cross-wave reduction into LDS (ds_add_f32) --------------------------
  const int mo = half * 8;
#pragma unroll
  for (int mi = 0; mi < 4; ++mi)
#pragma unroll
    for (int ni = 0; ni < 4; ++ni)
#pragma unroll
      for (int i = 0; i < 8; ++i)
        atomicAdd(&Ssum[mi*16 + mo + i][ni*16 + r16], acc[mi][ni][i]);
  __syncthreads();

  // ---- temperature + row softmax over 48 cols ------------------------------
  if (tid < ACH) {
    const float t = temperature[h];
    float mx = -3.402823466e+38f;
    for (int j = 0; j < ACH; ++j) mx = fmaxf(mx, Ssum[tid][j] * t);
    float s = 0.f;
    float* orow = attn_out + ((long long)bh * ACH + tid) * ACH;
    for (int j = 0; j < ACH; ++j) {
      float e = __expf(Ssum[tid][j] * t - mx);
      Ssum[tid][j] = e;
      s += e;
    }
    const float inv = 1.f / s;
    for (int j = 0; j < ACH; ++j) orow[j] = Ssum[tid][j] * inv;
  }
}

// ========== fuse: M_b[o][h*48+d] = sum_c proj_w[o][h*48+c]*attn[b,h][c][d] =
__global__ __launch_bounds__(256)
void fuse_proj_attn(const float* __restrict__ proj_w,  // [C][C]
                    const float* __restrict__ attn,    // [B][8][48][48]
                    float* __restrict__ Mmat,          // [B][C][C]
                    int C)
{
  const int idx = blockIdx.x * 256 + threadIdx.x;
  const int b = blockIdx.y;
  if (idx >= C * C) return;
  const int o = idx / C, g = idx % C;
  const int h = g / ACH, d = g % ACH;
  const float* pw = proj_w + (long long)o * C + h * ACH;
  const float* at = attn + ((long long)(b * 8 + h) * ACH) * ACH + d;
  float s = 0.f;
#pragma unroll 8
  for (int c = 0; c < ACH; ++c) s = fmaf(pw[c], at[(long long)c * ACH], s);
  Mmat[((long long)b * C + o) * C + g] = s;
}

// ============================== launcher ===================================
extern "C" void kernel_launch(void* const* d_in, const int* in_sizes, int n_in,
                              void* d_out, int out_size, void* d_ws, size_t ws_size,
                              hipStream_t stream)
{
  const float* x       = (const float*)d_in[0];  // [4][384][128][128]
  const float* qkv_w   = (const float*)d_in[1];  // [1152][384]
  const float* dw_w    = (const float*)d_in[2];  // [1152][1][3][3]
  const float* proj_w  = (const float*)d_in[3];  // [384][384]
  const float* temp    = (const float*)d_in[4];  // [8]
  float* out = (float*)d_out;                    // [4][384][128][128]

  const int B = 4, C = 384, H = 128, W = 128;
  const int C3 = 3 * C, HEADS = 8;
  constexpr int NN = 128 * 128;   // 16384
  constexpr int KC = 384;

  char* ws = (char*)d_ws;
  const size_t s_qkv = (size_t)B * C3 * NN * sizeof(float);   // ~302 MB
  float* qkv_raw = (float*)ws;
  float* qkv_dw  = (float*)(ws + s_qkv);
  float* ssq     = (float*)(ws + 2 * s_qkv);
  const size_t s_ssq = (size_t)B * 2 * C * sizeof(float);
  float* attn    = (float*)(ws + 2 * s_qkv + s_ssq);
  const size_t s_attn = (size_t)B * HEADS * ACH * ACH * sizeof(float);
  float* Mmat    = (float*)(ws + 2 * s_qkv + s_ssq + s_attn);

  hipMemsetAsync(ssq, 0, s_ssq, stream);

  // 1) qkv = qkv_w x x  (per batch: [1152x384] x [384x16384])
  {
    dim3 g(NN / BN, C3 / BM, B);
    gemm_bf16_wmma<NN, KC><<<g, 256, 0, stream>>>(qkv_w, 0LL,
                                                  x, (long long)C * NN,
                                                  qkv_raw, (long long)C3 * NN);
  }
  // 2) depthwise 3x3 + q/k sum-of-squares
  {
    dim3 g(W / 16, H / 16, B * C3);
    dwconv3x3_ssq<<<g, 256, 0, stream>>>(qkv_raw, dw_w, qkv_dw, ssq, C3, C, H, W);
  }
  // 3) attention scores + softmax, one block per (b, head)
  attn_softmax_wmma<<<B * HEADS, 256, 0, stream>>>(qkv_dw, ssq, temp, attn, C, NN);
  // 4) fold attn into projection: M_b = proj_w (.) attn_blockdiag
  {
    dim3 g((C * C + 255) / 256, B);
    fuse_proj_attn<<<g, 256, 0, stream>>>(proj_w, attn, Mmat, C);
  }
  // 5) out = M_b x v  (per batch: [384x384] x [384x16384])
  {
    dim3 g(NN / BN, C / BM, B);
    gemm_bf16_wmma<NN, KC><<<g, 256, 0, stream>>>(Mmat, (long long)C * C,
                                                  qkv_dw + (size_t)2 * C * NN, (long long)C3 * NN,
                                                  out, (long long)C * NN);
  }
}
